// WaveFunctionElectron_44513041056442
// MI455X (gfx1250) — matrix operand order repaired and checked by reading, not compiled, not run on hardware
//
#include <hip/hip_runtime.h>
#include <math.h>

// ---------------------------------------------------------------------------
// MI455X (gfx1250, wave32) implementation of the MoE transformer reference.
// All GEMMs run through V_WMMA_F32_16X16X32_BF16 (fp32 -> packed bf16 on the
// fly, fp32 accumulation). 128x128 block tile, 8 waves, 8 WMMA/wave/K-step.
// LDS is laid out in *fragment order* so every operand fragment is two
// 16B-aligned ds_load_b128 with immediate offsets from one base register.
// Double-buffered LDS panels + double-buffered B fragments + register
// prefetch of the next K panel.
// ---------------------------------------------------------------------------

typedef __attribute__((ext_vector_type(16))) __bf16 v16bf;
typedef __attribute__((ext_vector_type(2)))  __bf16 v2bf;
typedef __attribute__((ext_vector_type(8)))  float  v8f;

#define ACT_NONE 0
#define ACT_GELU 1

#if defined(__has_builtin)
#if __has_builtin(__builtin_amdgcn_cvt_pk_bf16_f32)
#define MOE_HAVE_CVT_PK_BF16 1
#endif
#endif

// round-to-nearest-even fp32 pair -> packed bf16x2 (lo = first element)
__device__ __forceinline__ unsigned int pack_bf16x2(float lo, float hi) {
#ifdef MOE_HAVE_CVT_PK_BF16
  union { v2bf v; unsigned int u; } r;
  r.v = __builtin_amdgcn_cvt_pk_bf16_f32(lo, hi);   // v_cvt_pk_bf16_f32
  return r.u;
#else
  unsigned int a = __float_as_uint(lo);
  unsigned int b = __float_as_uint(hi);
  unsigned int ra = (a + 0x7FFFu + ((a >> 16) & 1u)) >> 16;
  unsigned int rb = (b + 0x7FFFu + ((b >> 16) & 1u)) & 0xFFFF0000u;
  return ra | rb;
#endif
}

__device__ __forceinline__ float gelu_exact(float v) {
  return 0.5f * v * (1.0f + erff(v * 0.70710678118654752440f));
}

union Frag16 { v16bf v; unsigned int u32[8]; uint4 q[2]; };

// ---------------------------------------------------------------------------
// Generic GEMM: C(MxN) = post( A(MxK) @ W(KxN) + bias )
//   ACT: 0 none / 1 exact GELU;  RS: per-row scale (gate);  ACC: C += result.
// ---------------------------------------------------------------------------
template <int ACT, int RS, int ACC>
__global__ __launch_bounds__(256)
void moe_gemm_bf16_wmma(const float* __restrict__ A, const float* __restrict__ W,
                        const float* __restrict__ bias, float* __restrict__ C,
                        int M, int N, int K,
                        const float* __restrict__ rowscale, int rs_stride) {
  // A: [buf][row(128)][k(32)+pad] -> 80B rows (stride 20 dwords: bank-perfect)
  // B: [buf][col(16)][nt(8)][k(32)] + 16B col pad -> 528B cols (132 dwords:
  //    bank-perfect); all 8 fragments of a lane within 512B of one base.
  __shared__ __align__(16) unsigned short As[2][128][40];
  __shared__ __align__(16) unsigned short Bs[2][16][264];

  const int tid  = threadIdx.x;
  const int wave = tid >> 5;
  const int lane = tid & 31;
  const int M0 = blockIdx.y * 128;
  const int N0 = blockIdx.x * 128;

  v8f acc[8];
#pragma unroll
  for (int i = 0; i < 8; ++i)
#pragma unroll
    for (int j = 0; j < 8; ++j) acc[i][j] = 0.0f;

  const int arow  = wave * 16 + (lane & 15);
  const int col0  = lane & 15;
  const int khalf = lane >> 4;           // which K half this lane holds

  float2 aR[8];                          // staged A pairs (K-consecutive)
  float  bR[16];                         // staged B pairs (K-consecutive)

  // global -> register prefetch (pair index f = tid + i*256 covers the tile)
  auto loadA = [&](int k0) {
#pragma unroll
    for (int i = 0; i < 8; ++i) {
      int f = tid + i * 256;             // 128 rows x 16 K-pairs
      int row = f >> 4, kp = f & 15;
      aR[i] = *(const float2*)&A[(size_t)(M0 + row) * K + k0 + 2 * kp];
    }
  };
  auto loadB = [&](int k0) {
#pragma unroll
    for (int i = 0; i < 8; ++i) {
      int f = tid + i * 256;             // 16 K-pairs x 128 cols
      int n = N0 + (f & 127), kk = (f >> 7) * 2;
      bR[2 * i]     = (n < N) ? W[(size_t)(k0 + kk)     * N + n] : 0.0f;
      bR[2 * i + 1] = (n < N) ? W[(size_t)(k0 + kk + 1) * N + n] : 0.0f;
    }
  };
  // register -> LDS (packed bf16x2, one b32 store per pair, fragment order)
  auto storeTile = [&](int p) {
    char* ab = (char*)&As[p][0][0];
    char* bb = (char*)&Bs[p][0][0];
#pragma unroll
    for (int i = 0; i < 8; ++i) {
      int f = tid + i * 256;
      int row = f >> 4, kp = f & 15;
      *(unsigned int*)(ab + row * 80 + kp * 4) = pack_bf16x2(aR[i].x, aR[i].y);
    }
#pragma unroll
    for (int i = 0; i < 8; ++i) {
      int f = tid + i * 256;
      int n = f & 127, kk = (f >> 7) * 2;
      *(unsigned int*)(bb + (n & 15) * 528 + (n >> 4) * 64 + kk * 2) =
          pack_bf16x2(bR[2 * i], bR[2 * i + 1]);
    }
  };

  loadA(0);
  loadB(0);
  storeTile(0);
  __syncthreads();

  const int nsteps = K >> 5;
  for (int s = 0; s < nsteps; ++s) {
    const int p = s & 1;
    const bool more = (s + 1 < nsteps);

    // issue next panel's global loads first (latency hidden by WMMAs below)
    if (more) { loadA((s + 1) * 32); loadB((s + 1) * 32); }

    // A fragment: two 16B chunks at one base (immediate offsets)
    const char* abase = (const char*)&As[p][0][0] + arow * 80 + khalf * 16;
    Frag16 a;
    a.q[0] = *(const uint4*)(abase);
    a.q[1] = *(const uint4*)(abase + 32);

    // 8 WMMA with double-buffered B fragments; every fragment is two
    // immediate-offset ds_load_b128 from a single per-lane base register.
    const char* bbase = (const char*)&Bs[p][0][0] + col0 * 528 + khalf * 16;
    Frag16 bf[2];
    bf[0].q[0] = *(const uint4*)(bbase);
    bf[0].q[1] = *(const uint4*)(bbase + 32);
#pragma unroll
    for (int nt = 0; nt < 8; ++nt) {
      if (nt < 7) {
        bf[(nt + 1) & 1].q[0] = *(const uint4*)(bbase + (nt + 1) * 64);
        bf[(nt + 1) & 1].q[1] = *(const uint4*)(bbase + (nt + 1) * 64 + 32);
      }
      acc[nt] = __builtin_amdgcn_wmma_f32_16x16x32_bf16(
          false, a.v, false, bf[nt & 1].v, (short)0, acc[nt], false, false);
    }

    // stage next panel into the opposite LDS buffer, single barrier per panel
    if (more) storeTile(p ^ 1);
    __syncthreads();
  }

  // ---- epilogue: C/D layout -> col = lane&15, row = vgpr + 8*(lane>>4) ----
  const int rbase = wave * 16 + khalf * 8;
#pragma unroll
  for (int nt = 0; nt < 8; ++nt) {
    int col = N0 + nt * 16 + col0;
    if (col >= N) continue;
    float bv = bias[col];
#pragma unroll
    for (int r = 0; r < 8; ++r) {
      int row = M0 + rbase + r;
      float v = acc[nt][r] + bv;
      if (ACT == ACT_GELU) v = gelu_exact(v);
      if (RS) v *= rowscale[(size_t)row * rs_stride];
      size_t o = (size_t)row * N + col;
      if (ACC) C[o] += v; else C[o] = v;
    }
  }
}

// ---------------------------------------------------------------------------
// Embedding gather: out[b,s,:] = emb[s, x[b,s], :]   emb: (S,2,E)
// ---------------------------------------------------------------------------
__global__ __launch_bounds__(128)
void moe_embed(const float* __restrict__ emb, const int* __restrict__ x,
               float* __restrict__ out) {
  const int E = 512, S = 64;
  int t = blockIdx.x;                 // token = b*S + s
  int s = t & (S - 1);
  int id = x[t];
  const float* src = emb + ((size_t)(s * 2 + id)) * E;
  float* dst = out + (size_t)t * E;
  for (int e = threadIdx.x; e < E; e += 128) dst[e] = src[e];
}

// ---------------------------------------------------------------------------
// Causal attention, one block per (q, h, b). S=64, HD=64, qkv: (B,S,3E).
// ---------------------------------------------------------------------------
__global__ __launch_bounds__(64)
void moe_attn(const float* __restrict__ qkv, float* __restrict__ o) {
  const int S = 64, E = 512, T3 = 1536, HD = 64;
  int q = blockIdx.x, h = blockIdx.y, b = blockIdx.z;
  int tid = threadIdx.x;

  __shared__ float qs[64];
  __shared__ float sc[64];
  __shared__ float red[64];

  qs[tid] = qkv[((size_t)(b * S + q)) * T3 + h * HD + tid];
  __syncthreads();

  const float* krow = qkv + ((size_t)(b * S + tid)) * T3 + E + h * HD;
  float s = 0.0f;
#pragma unroll 8
  for (int d = 0; d < HD; ++d) s += qs[d] * krow[d];
  s *= 0.125f;                        // 1/sqrt(64)
  bool valid = (tid <= q);
  red[tid] = valid ? s : -3.0e38f;
  __syncthreads();
  for (int off = 32; off > 0; off >>= 1) {
    if (tid < off) red[tid] = fmaxf(red[tid], red[tid + off]);
    __syncthreads();
  }
  float mx = red[0];
  __syncthreads();
  float e = valid ? expf(s - mx) : 0.0f;
  sc[tid] = e;
  red[tid] = e;
  __syncthreads();
  for (int off = 32; off > 0; off >>= 1) {
    if (tid < off) red[tid] += red[tid + off];
    __syncthreads();
  }
  float inv = 1.0f / red[0];

  const float* vbase = qkv + (size_t)(b * S) * T3 + 2 * E + h * HD;
  float acc = 0.0f;
  for (int j = 0; j <= q; ++j) acc += sc[j] * vbase[(size_t)j * T3 + tid];
  o[((size_t)(b * S + q)) * E + h * HD + tid] = acc * inv;
}

// ---------------------------------------------------------------------------
// LayerNorm over E=512 (optional fused residual add). One block per token.
// ---------------------------------------------------------------------------
__global__ __launch_bounds__(128)
void moe_ln(const float* __restrict__ xin, const float* __restrict__ resid,
            const float* __restrict__ w, const float* __restrict__ b,
            float* __restrict__ out) {
  const int E = 512;
  int m = blockIdx.x, tid = threadIdx.x;
  const float* xr = xin + (size_t)m * E;
  const float* rr = resid ? resid + (size_t)m * E : nullptr;
  __shared__ float red[128];

  float v[4]; float s = 0.0f;
#pragma unroll
  for (int i = 0; i < 4; ++i) {
    int e = tid + i * 128;
    v[i] = xr[e] + (rr ? rr[e] : 0.0f);
    s += v[i];
  }
  red[tid] = s; __syncthreads();
  for (int off = 64; off > 0; off >>= 1) {
    if (tid < off) red[tid] += red[tid + off];
    __syncthreads();
  }
  float mean = red[0] * (1.0f / 512.0f);
  __syncthreads();
  float vs = 0.0f;
#pragma unroll
  for (int i = 0; i < 4; ++i) { float d = v[i] - mean; vs += d * d; }
  red[tid] = vs; __syncthreads();
  for (int off = 64; off > 0; off >>= 1) {
    if (tid < off) red[tid] += red[tid + off];
    __syncthreads();
  }
  float rstd = rsqrtf(red[0] * (1.0f / 512.0f) + 1e-5f);
  __syncthreads();
#pragma unroll
  for (int i = 0; i < 4; ++i) {
    int e = tid + i * 128;
    out[(size_t)m * E + e] = (v[i] - mean) * rstd * w[e] + b[e];
  }
}

// ---------------------------------------------------------------------------
// Routing: sim = softmax(x @ centroid^T); top-2 of (sim + bias); scatter gate
// renormalized over the selected pair. One block (8 waves) per token.
// ---------------------------------------------------------------------------
__global__ __launch_bounds__(256)
void moe_route(const float* __restrict__ x, const float* __restrict__ centroid,
               const float* __restrict__ bias, float* __restrict__ gate) {
  const int E = 512, R = 8;
  int m = blockIdx.x;
  int wave = threadIdx.x >> 5, lane = threadIdx.x & 31;
  const float* xr = x + (size_t)m * E;
  const float* cr = centroid + (size_t)wave * E;
  float s = 0.0f;
  for (int k = lane; k < E; k += 32) s += xr[k] * cr[k];
#pragma unroll
  for (int off = 16; off > 0; off >>= 1) s += __shfl_down(s, off, 32);
  __shared__ float sims[8];
  if (lane == 0) sims[wave] = s;
  __syncthreads();
  if (threadIdx.x == 0) {
    float mx = sims[0];
    for (int e = 1; e < R; ++e) mx = fmaxf(mx, sims[e]);
    float p[8]; float sum = 0.0f;
    for (int e = 0; e < R; ++e) { p[e] = expf(sims[e] - mx); sum += p[e]; }
    float inv = 1.0f / sum;
    for (int e = 0; e < R; ++e) p[e] *= inv;
    int i0 = 0; float b0 = -3.0e38f;
    for (int e = 0; e < R; ++e) { float t = p[e] + bias[e]; if (t > b0) { b0 = t; i0 = e; } }
    int i1 = 0; float b1 = -3.0e38f;
    for (int e = 0; e < R; ++e) { if (e == i0) continue; float t = p[e] + bias[e]; if (t > b1) { b1 = t; i1 = e; } }
    float g = 1.0f / (p[i0] + p[i1]);
    for (int e = 0; e < R; ++e) gate[(size_t)m * R + e] = 0.0f;
    gate[(size_t)m * R + i0] = p[i0] * g;
    gate[(size_t)m * R + i1] = p[i1] * g;
  }
}

__global__ __launch_bounds__(256)
void moe_copy(float* __restrict__ dst, const float* __restrict__ src, size_t n) {
  size_t i = (size_t)blockIdx.x * blockDim.x + threadIdx.x;
  size_t stride = (size_t)gridDim.x * blockDim.x;
  for (; i < n; i += stride) dst[i] = src[i];
}

// ---------------------------------------------------------------------------
// Host-side dispatch over the (ACT, RS, ACC) combinations actually used.
// ---------------------------------------------------------------------------
static inline void launch_gemm(const float* A, const float* W, const float* bias,
                               float* C, int M, int N, int K, int act,
                               const float* rowscale, int rs_stride, int accumulate,
                               hipStream_t s) {
  dim3 grid((N + 127) / 128, M / 128);
  if (act == ACT_GELU) {
    moe_gemm_bf16_wmma<ACT_GELU, 0, 0><<<grid, 256, 0, s>>>(A, W, bias, C, M, N, K, nullptr, 0);
  } else if (rowscale) {
    moe_gemm_bf16_wmma<ACT_NONE, 1, 1><<<grid, 256, 0, s>>>(A, W, bias, C, M, N, K, rowscale, rs_stride);
  } else if (accumulate) {
    moe_gemm_bf16_wmma<ACT_NONE, 0, 1><<<grid, 256, 0, s>>>(A, W, bias, C, M, N, K, nullptr, 0);
  } else {
    moe_gemm_bf16_wmma<ACT_NONE, 0, 0><<<grid, 256, 0, s>>>(A, W, bias, C, M, N, K, nullptr, 0);
  }
}

extern "C" void kernel_launch(void* const* d_in, const int* in_sizes, int n_in,
                              void* d_out, int out_size, void* d_ws, size_t ws_size,
                              hipStream_t stream) {
  (void)in_sizes; (void)n_in; (void)out_size; (void)ws_size;
  const int B = 128, S = 64, E = 512, FF = 2048, ROUTED = 8, DEPTH = 4;
  const int M = B * S;                    // 8192 tokens

  // ---- unpack inputs (JAX pytree flattening: dict keys sorted per level) ----
  int p = 0;
  auto nf = [&]() { return (const float*)d_in[p++]; };
  const float* emb = nf();
  struct LayerP {
    const float *bias, *bo, *bqkv, *centroid, *ln1b, *ln1w, *ln2b, *ln2w;
    const float *r_b1[8], *r_b2[8], *r_w1[8], *r_w2[8];
    const float *s_b1, *s_b2, *s_w1, *s_w2;
    const float *wo, *wqkv;
  } L[DEPTH];
  for (int l = 0; l < DEPTH; ++l) {
    L[l].bias = nf(); L[l].bo = nf(); L[l].bqkv = nf(); L[l].centroid = nf();
    L[l].ln1b = nf(); L[l].ln1w = nf(); L[l].ln2b = nf(); L[l].ln2w = nf();
    for (int e = 0; e < ROUTED; ++e) {
      L[l].r_b1[e] = nf(); L[l].r_b2[e] = nf();
      L[l].r_w1[e] = nf(); L[l].r_w2[e] = nf();
    }
    L[l].s_b1 = nf(); L[l].s_b2 = nf(); L[l].s_w1 = nf(); L[l].s_w2 = nf();
    L[l].wo = nf(); L[l].wqkv = nf();
  }
  const float* t_b1 = nf(); const float* t_b2 = nf();
  const float* t_w1 = nf(); const float* t_w2 = nf();
  const int* x = (const int*)d_in[p++];

  // ---- workspace layout (floats) ----
  float* X   = (float*)d_ws;              // M*E      activations
  float* Y   = X   + (size_t)M * E;       // M*E      residual accumulator
  float* QKV = Y   + (size_t)M * E;       // M*3E
  float* O   = QKV + (size_t)M * 3 * E;   // M*E      attention output
  float* Hb  = O   + (size_t)M * E;       // M*FF     FFN hidden
  float* G   = Hb  + (size_t)M * FF;      // M*ROUTED gates

  // ---- embedding ----
  moe_embed<<<M, 128, 0, stream>>>(emb, x, X);

  for (int l = 0; l < DEPTH; ++l) {
    const LayerP& lp = L[l];
    // qkv = x @ wqkv + bqkv
    launch_gemm(X, lp.wqkv, lp.bqkv, QKV, M, 3 * E, E, ACT_NONE, nullptr, 0, 0, stream);
    // causal attention per (q,h,b)
    moe_attn<<<dim3(S, 8, B), 64, 0, stream>>>(QKV, O);
    // o = attn @ wo + bo   (into Y as temp)
    launch_gemm(O, lp.wo, lp.bo, Y, M, E, E, ACT_NONE, nullptr, 0, 0, stream);
    // x = ln1(x + o)
    moe_ln<<<M, 128, 0, stream>>>(X, Y, lp.ln1w, lp.ln1b, X);
    // y = x
    moe_copy<<<2048, 256, 0, stream>>>(Y, X, (size_t)M * E);
    // shared expert: y += w2(gelu(x@w1+b1)) + b2
    launch_gemm(X, lp.s_w1, lp.s_b1, Hb, M, FF, E, ACT_GELU, nullptr, 0, 0, stream);
    launch_gemm(Hb, lp.s_w2, lp.s_b2, Y, M, E, FF, ACT_NONE, nullptr, 0, 1, stream);
    // routing gates
    moe_route<<<M, 256, 0, stream>>>(X, lp.centroid, lp.bias, G);
    // routed experts (dense, gated accumulate — matches reference exactly)
    for (int e = 0; e < ROUTED; ++e) {
      launch_gemm(X, lp.r_w1[e], lp.r_b1[e], Hb, M, FF, E, ACT_GELU, nullptr, 0, 0, stream);
      launch_gemm(Hb, lp.r_w2[e], lp.r_b2[e], Y, M, E, FF, ACT_NONE, G + e, ROUTED, 1, stream);
    }
    // x = ln2(y)
    moe_ln<<<M, 128, 0, stream>>>(Y, nullptr, lp.ln2w, lp.ln2b, X);
  }

  // ---- tail: Linear -> GELU -> Linear(4) ----
  launch_gemm(X, t_w1, t_b1, Hb, M, FF, E, ACT_GELU, nullptr, 0, 0, stream);
  launch_gemm(Hb, t_w2, t_b2, (float*)d_out, M, 4, FF, ACT_NONE, nullptr, 0, 0, stream);
}
